// GaussiansGenerator_67989332296093
// MI455X (gfx1250) — matrix operand order repaired and checked by reading, not compiled, not run on hardware
//
#include <hip/hip_runtime.h>
#include <hip/hip_bf16.h>
#include <math.h>

typedef _Float16 half_t;
typedef __attribute__((ext_vector_type(16))) half_t v16h;
typedef __attribute__((ext_vector_type(8)))  float  v8f;
typedef unsigned int uint32;

// ---------------------------------------------------------------------------
// WMMA 16x16x32 f16 lane layouts (CDNA5 ISA):
//   A (16x32): lanes 0-15 rows M=0..15, K=ks..ks+15 (this code's k order, not the ISA table);
//              lanes 16-31 same rows, K=ks+16..31
//   B (32x16): lanes 0-15 col N=lane&15, K=ks..ks+15; lanes 16-31 K=ks+16..31 (same k order as A)
//   C/D f32  : VGPR r -> M=r (lanes 0-15) / M=r+8 (lanes 16-31), N=lane&15
// Weights are pre-swizzled into the exact B lane layout so a lane's 16
// halves are 32 contiguous bytes -> two global_load_b128 per k-step.
// Packed layout: Wp[(((kstep*NT + ntile)*32 + lane)*16 + i]
// ---------------------------------------------------------------------------

// pre-swizzle f32 weights [K+rowOff, NOUT] (rows rowOff..rowOff+K-1) into
// f16 WMMA-B fragment order.
__global__ void pack_w_kernel(const float* __restrict__ W, half_t* __restrict__ Wp,
                              int K, int NOUT, int rowOff) {
    int f = blockIdx.x * blockDim.x + threadIdx.x;
    if (f >= K * NOUT) return;
    int i    = f & 15;
    int lane = (f >> 4) & 31;
    int rest = f >> 9;
    int NT   = NOUT >> 4;
    int nt   = rest % NT;
    int ks   = rest / NT;                       // k-step index
    int k    = ks * 32 + (lane >> 4) * 16 + i;  // K index within packed rows
    int col  = nt * 16 + (lane & 15);
    Wp[f] = (half_t)W[(size_t)(k + rowOff) * NOUT + col];
}

// Generic fused GEMM: out = act(A[M,K] @ W + bias) (+ residual)
// A f16 row-major (staged per-block in LDS); W pre-packed f16 fragments.
// Block: (NOUT/16) waves, each wave one 16-col tile; block covers 16 rows.
// ACT: 0=none 1=relu 2=leaky_relu(0.2)
template<int ACT, bool RESADD, bool WRITE_H>
__global__ void gemm_wmma_kernel(const half_t* __restrict__ A,
                                 const half_t* __restrict__ Wp,   // packed
                                 const float*  __restrict__ bias,
                                 const float*  __restrict__ res,
                                 float*        __restrict__ outF,
                                 half_t*       __restrict__ outH,
                                 int M, int K, int NOUT) {
    extern __shared__ half_t sA[];            // 16 x K halves
    const int m0   = blockIdx.x * 16;
    const int lane = threadIdx.x & 31;
    const int wave = threadIdx.x >> 5;
    const int Kd   = K >> 1;
    const int NT   = NOUT >> 4;

    // cooperative stage of the 16xK A tile into LDS
    const uint32* Au  = (const uint32*)A;
    uint32*       sAu = (uint32*)sA;
    for (int idx = threadIdx.x; idx < 16 * Kd; idx += blockDim.x) {
        int r  = idx / Kd;
        int kd = idx - r * Kd;
        int rr = m0 + r; if (rr >= M) rr = M - 1;
        sAu[idx] = Au[(size_t)rr * Kd + kd];
    }
    __syncthreads();

    const int mloc = lane & 15;
    const int hi   = lane >> 4;
    const int ncol = wave * 16 + mloc;
    const v16h* Bp = (const v16h*)Wp;

    v8f acc = {};
    for (int ks = 0; ks < K; ks += 32) {
        union { v16h v; uint32 u[8]; } a;
        const uint32* ap = sAu + mloc * Kd + ((ks + hi * 16) >> 1);
#pragma unroll
        for (int j = 0; j < 4; ++j) { a.u[j] = ap[j]; a.u[4 + j] = ap[4 + j]; }

        v16h bv = Bp[((size_t)(ks >> 5) * NT + wave) * 32 + lane];

        acc = __builtin_amdgcn_wmma_f32_16x16x32_f16(false, a.v, false, bv,
                                                     (short)0, acc, false, false);
    }

    const float bb = bias[ncol];
#pragma unroll
    for (int r = 0; r < 8; ++r) {
        int row = m0 + r + hi * 8;
        if (row >= M) continue;
        float v = acc[r] + bb;
        if (ACT == 1)      v = fmaxf(v, 0.0f);
        else if (ACT == 2) v = (v > 0.0f) ? v : 0.2f * v;
        if (RESADD)        v += res[(size_t)row * NOUT + ncol];
        outF[(size_t)row * NOUT + ncol] = v;
        if (WRITE_H) outH[(size_t)row * NOUT + ncol] = (half_t)v;
    }
}

// Edge message kernel: 16 edges/block,
//   m = relu([pos_j - pos_i + delta_i, x_j] @ Wf + bf); atomicAdd agg[dst]
// x-part (128 rows of Wf, pre-packed) via WMMA; 3 pos rows via f32 FMA.
// Padded edges scatter into sink row `sinkRow` (= N) -> branch-free epilogue.
__global__ void edge_f_kernel(const half_t* __restrict__ xh,     // [N,128] f16
                              const float*  __restrict__ posn,   // [N,3]
                              const float*  __restrict__ delta,  // [N,3]
                              const int*    __restrict__ srcI,
                              const int*    __restrict__ dstI,
                              const half_t* __restrict__ Wfp,    // packed rows 3..130
                              const float*  __restrict__ Wf,     // f32 [131,128] (pos rows)
                              const float*  __restrict__ bf,     // [128]
                              float*        __restrict__ agg,    // [N+1,128]
                              int E, int sinkRow) {
    __shared__ int    ssrc[16];
    __shared__ int    sdst[16];
    __shared__ float  spe[16][3];
    __shared__ half_t sA[16 * 128];

    const int e0 = blockIdx.x * 16;
    const int t  = threadIdx.x;
    if (t < 16) {
        int e = e0 + t;
        if (e < E) {
            int s = srcI[e], d = dstI[e];
            ssrc[t] = s; sdst[t] = d;
#pragma unroll
            for (int c = 0; c < 3; ++c)
                spe[t][c] = posn[s * 3 + c] - posn[d * 3 + c] + delta[d * 3 + c];
        } else {
            ssrc[t] = 0; sdst[t] = sinkRow;          // write-only sink row
            spe[t][0] = spe[t][1] = spe[t][2] = 0.0f;
        }
    }
    __syncthreads();

    // gather the 16 source-node rows into LDS (shared by all 8 waves)
    const uint32* xu  = (const uint32*)xh;
    uint32*       sAu = (uint32*)sA;
    for (int idx = t; idx < 16 * 64; idx += blockDim.x) {
        int r  = idx >> 6;
        int kd = idx & 63;
        sAu[idx] = xu[(size_t)ssrc[r] * 64 + kd];
    }
    __syncthreads();

    const int lane = t & 31;
    const int wave = t >> 5;
    const int mloc = lane & 15;
    const int hi   = lane >> 4;
    const int ncol = wave * 16 + mloc;
    const v16h* Bp = (const v16h*)Wfp;

    v8f acc = {};
#pragma unroll
    for (int ks = 0; ks < 128; ks += 32) {
        union { v16h v; uint32 u[8]; } a;
        const uint32* ap = sAu + mloc * 64 + ((ks + hi * 16) >> 1);
#pragma unroll
        for (int j = 0; j < 4; ++j) { a.u[j] = ap[j]; a.u[4 + j] = ap[4 + j]; }

        v16h bv = Bp[((size_t)(ks >> 5) * 8 + wave) * 32 + lane];

        acc = __builtin_amdgcn_wmma_f32_16x16x32_f16(false, a.v, false, bv,
                                                     (short)0, acc, false, false);
    }

    const float bb = bf[ncol];
    const float w0 = Wf[ncol];
    const float w1 = Wf[128 + ncol];
    const float w2 = Wf[256 + ncol];
#pragma unroll
    for (int r = 0; r < 8; ++r) {
        int el = r + hi * 8;                 // local edge index
        float v = acc[r] + bb + spe[el][0] * w0 + spe[el][1] * w1 + spe[el][2] * w2;
        v = fmaxf(v, 0.0f);
        atomicAdd(&agg[(size_t)sdst[el] * 128 + ncol], v);
    }
}

// Gaussian encoding: x = [cos(2pi*pos@B^T), sin(2pi*pos@B^T)]  -> f32 + f16
__global__ void encode_kernel(const float* __restrict__ pos,
                              const float* __restrict__ Bm,   // [64,3]
                              float* __restrict__ xf, half_t* __restrict__ xh, int N) {
    int i = blockIdx.x * blockDim.x + threadIdx.x;
    if (i >= N) return;
    float p0 = pos[i * 3], p1 = pos[i * 3 + 1], p2 = pos[i * 3 + 2];
    const float TWO_PI = 6.283185307179586f;
#pragma unroll 4
    for (int j = 0; j < 64; ++j) {
        float vv = TWO_PI * (p0 * Bm[j * 3] + p1 * Bm[j * 3 + 1] + p2 * Bm[j * 3 + 2]);
        float c = cosf(vv), s = sinf(vv);
        xf[(size_t)i * 128 + j] = c;         xf[(size_t)i * 128 + 64 + j] = s;
        xh[(size_t)i * 128 + j] = (half_t)c; xh[(size_t)i * 128 + 64 + j] = (half_t)s;
    }
}

// small head: out3 = tanh(A[N,K] @ W[K,3] + b)
__global__ void lin3_tanh_kernel(const float* __restrict__ A,
                                 const float* __restrict__ W,
                                 const float* __restrict__ b,
                                 float* __restrict__ out, int N, int K) {
    int i = blockIdx.x * blockDim.x + threadIdx.x;
    if (i >= N) return;
    float s0 = b[0], s1 = b[1], s2 = b[2];
    const float* a = A + (size_t)i * K;
    for (int k = 0; k < K; ++k) {
        float av = a[k];
        s0 += av * W[k * 3]; s1 += av * W[k * 3 + 1]; s2 += av * W[k * 3 + 2];
    }
    out[i * 3] = tanhf(s0); out[i * 3 + 1] = tanhf(s1); out[i * 3 + 2] = tanhf(s2);
}

// column-max pooling, stage 1
__global__ void colmax_partial_kernel(const float* __restrict__ x,
                                      float* __restrict__ part, int N) {
    int col = threadIdx.x;     // 128 threads
    int b   = blockIdx.x;
    float m = -__builtin_inff();
    for (int i = b; i < N; i += gridDim.x)
        m = fmaxf(m, x[(size_t)i * 128 + col]);
    part[b * 128 + col] = m;
}

// stage 2: final max + pg_global linear + leaky_relu
__global__ void colmax_final_pg_kernel(const float* __restrict__ part, int PARTS,
                                       const float* __restrict__ Wpg,
                                       const float* __restrict__ bpg,
                                       float* __restrict__ hglob) {
    __shared__ float h[128];
    int j = threadIdx.x;       // 128 threads, 1 block
    float m = -__builtin_inff();
    for (int b = 0; b < PARTS; ++b) m = fmaxf(m, part[b * 128 + j]);
    h[j] = m;
    __syncthreads();
    float s = bpg[j];
    for (int k = 0; k < 128; ++k) s += h[k] * Wpg[k * 128 + j];
    hglob[j] = (s > 0.0f) ? s : 0.2f * s;
}

// xcat = f16 concat([x, broadcast(hglob)])  [N,256]
__global__ void concat_kernel(const float* __restrict__ xf,
                              const float* __restrict__ hglob,
                              half_t* __restrict__ xcat, int N) {
    size_t i = (size_t)blockIdx.x * blockDim.x + threadIdx.x;
    if (i >= (size_t)N * 256) return;
    int col = (int)(i & 255);
    size_t row = i >> 8;
    float v = (col < 128) ? xf[row * 128 + col] : hglob[col - 128];
    xcat[i] = (half_t)v;
}

__global__ void cvt_f32_f16_kernel(const float* __restrict__ a,
                                   half_t* __restrict__ o, int n) {
    int i = blockIdx.x * blockDim.x + threadIdx.x;
    if (i < n) o[i] = (half_t)a[i];
}

// ---------------------------------------------------------------------------

extern "C" void kernel_launch(void* const* d_in, const int* in_sizes, int n_in,
                              void* d_out, int out_size, void* d_ws, size_t ws_size,
                              hipStream_t stream) {
    const int N = in_sizes[0] / 3;
    const int E = in_sizes[1] / 2;

    const float* pos  = (const float*)d_in[0];
    const int*   ei   = (const int*)d_in[1];
    const int*   srcI = ei;
    const int*   dstI = ei + E;
    const float* Bm   = (const float*)d_in[3];

    // param leaf indices (setup_inputs insertion order, recursive)
    const int C1 = 4, C2 = 14, PGW = 24, PGB = 25, T1W = 26, T1B = 27,
              T2W = 28, T2B = 29, GGW = 30, GGB = 31, B1 = 32, B2 = 42;

    // ---- workspace carving (deterministic) ----
    char* wp = (char*)d_ws;
    auto alloc = [&](size_t bytes) -> void* {
        void* p = (void*)wp;
        wp += (bytes + 255) & ~(size_t)255;
        return p;
    };
    half_t* xh    = (half_t*)alloc((size_t)N * 128 * 2);
    float*  xf    = (float*) alloc((size_t)N * 128 * 4);
    float*  t0    = (float*) alloc((size_t)N * 128 * 4);
    half_t* t0h   = (half_t*)alloc((size_t)N * 128 * 2);
    float*  agg   = (float*) alloc((size_t)(N + 1) * 128 * 4);   // +1 sink row
    half_t* aggh  = (half_t*)alloc((size_t)N * 128 * 2);
    float*  delta = (float*) alloc((size_t)N * 3 * 4);
    half_t* xcat  = (half_t*)alloc((size_t)N * 256 * 2);
    float*  t64   = (float*) alloc((size_t)N * 64 * 4);
    const int PARTS = 256;
    float*  part  = (float*) alloc((size_t)PARTS * 128 * 4);
    float*  hglob = (float*) alloc(128 * 4);

    // pre-swizzle a weight leaf into WMMA-B fragment layout (f16)
    auto packW = [&](int idx, int K, int NOUT, int rowOff) -> half_t* {
        int n = K * NOUT;
        half_t* p = (half_t*)alloc((size_t)n * 2);
        pack_w_kernel<<<(n + 255) / 256, 256, 0, stream>>>(
            (const float*)d_in[idx], p, K, NOUT, rowOff);
        return p;
    };

    const int MG = (N + 15) / 16;                    // GEMM row-tile grid

    // one PointGNN block
    auto run_block = [&](int pb, const float* posp, float* outF) {
        half_t* Wh0 = packW(pb + 0, 128, 128, 0);
        half_t* Wfp = packW(pb + 4, 128, 128, 3);    // x-rows of Wf
        half_t* Wg0 = packW(pb + 6, 128, 128, 0);
        half_t* Wg1 = packW(pb + 8, 128, 128, 0);

        // delta = tanh(h1(relu(h0(x))))
        gemm_wmma_kernel<1, false, false><<<MG, 256, 16 * 128 * 2, stream>>>(
            xh, Wh0, (const float*)d_in[pb + 1], nullptr, t0, nullptr, N, 128, 128);
        lin3_tanh_kernel<<<(N + 127) / 128, 128, 0, stream>>>(
            t0, (const float*)d_in[pb + 2], (const float*)d_in[pb + 3], delta, N, 128);

        // agg = segment_sum(relu(f([pe, x_j])))
        hipMemsetAsync(agg, 0, (size_t)(N + 1) * 128 * 4, stream);
        edge_f_kernel<<<(E + 15) / 16, 256, 0, stream>>>(
            xh, posp, delta, srcI, dstI, Wfp,
            (const float*)d_in[pb + 4], (const float*)d_in[pb + 5], agg, E, N);

        // out = x + relu(g1(relu(g0(agg))))
        cvt_f32_f16_kernel<<<(N * 128 + 255) / 256, 256, 0, stream>>>(agg, aggh, N * 128);
        gemm_wmma_kernel<1, false, true><<<MG, 256, 16 * 128 * 2, stream>>>(
            aggh, Wg0, (const float*)d_in[pb + 7], nullptr, t0, t0h, N, 128, 128);
        gemm_wmma_kernel<1, true, true><<<MG, 256, 16 * 128 * 2, stream>>>(
            t0h, Wg1, (const float*)d_in[pb + 9], xf, outF, xh, N, 128, 128);
    };

    // ---- forward pass ----
    encode_kernel<<<(N + 127) / 128, 128, 0, stream>>>(pos, Bm, xf, xh, N);

    run_block(C1, pos, xf);                          // conv1
    run_block(C2, pos, xf);                          // conv2

    // global max pool + pg_global + lrelu
    colmax_partial_kernel<<<PARTS, 128, 0, stream>>>(xf, part, N);
    colmax_final_pg_kernel<<<1, 128, 0, stream>>>(
        part, PARTS, (const float*)d_in[PGW], (const float*)d_in[PGB], hglob);

    // xcat = [x, h]  (f16)
    concat_kernel<<<(int)(((size_t)N * 256 + 255) / 256), 256, 0, stream>>>(xf, hglob, xcat, N);

    // pos2 = tanh(tail2(lrelu(tail1(xcat))))  -> second output region
    half_t* Wt1 = packW(T1W, 256, 64, 0);
    gemm_wmma_kernel<2, false, false><<<MG, 128, 16 * 256 * 2, stream>>>(
        xcat, Wt1, (const float*)d_in[T1B], nullptr, t64, nullptr, N, 256, 64);
    float* pos2 = (float*)d_out + (size_t)N * 128;
    lin3_tanh_kernel<<<(N + 127) / 128, 128, 0, stream>>>(
        t64, (const float*)d_in[T2W], (const float*)d_in[T2B], pos2, N, 64);

    // x = lrelu(gg_global(xcat))
    half_t* Wgg = packW(GGW, 256, 128, 0);
    gemm_wmma_kernel<2, false, true><<<MG, 256, 16 * 256 * 2, stream>>>(
        xcat, Wgg, (const float*)d_in[GGB], nullptr, xf, xh, N, 256, 128);

    run_block(B1, pos2, xf);                         // block1
    run_block(B2, pos2, (float*)d_out);              // block2 -> first output region
}
